// Attention_56745107915026
// MI455X (gfx1250) — matrix operand order repaired and checked by reading, not compiled, not run on hardware
//
#include <hip/hip_runtime.h>

typedef __attribute__((ext_vector_type(4)))  int       v4i;
typedef __attribute__((ext_vector_type(4)))  float     v4f;
typedef __attribute__((ext_vector_type(8)))  float     v8f;
typedef __attribute__((ext_vector_type(8)))  _Float16  v8h;
typedef __attribute__((ext_vector_type(16))) _Float16  v16h;

// ---------------- tiling constants ----------------
constexpr int BM  = 128;   // block tile M
constexpr int BN  = 128;   // block tile N
constexpr int BK  = 32;    // K step == WMMA K
constexpr int LDK = 40;    // padded LDS row stride (halves); 80B rows stay 16B aligned

// -------- CDNA5 async global->LDS copies (guarded; sync fallback) --------
#if defined(__has_builtin)
#if __has_builtin(__builtin_amdgcn_global_load_async_to_lds_b128) && \
    __has_builtin(__builtin_amdgcn_s_wait_asynccnt)
#define USE_ASYNC_LDS 1
#endif
#endif
#ifndef USE_ASYNC_LDS
#define USE_ASYNC_LDS 0
#endif

using g_v4i = __attribute__((address_space(1))) v4i;   // global (AS1) 128-bit chunk
using l_v4i = __attribute__((address_space(3))) v4i;   // LDS    (AS3) 128-bit chunk

__device__ __forceinline__ void copy16_g2l(const _Float16* g, _Float16* l) {
#if USE_ASYNC_LDS
  __builtin_amdgcn_global_load_async_to_lds_b128((g_v4i*)g, (l_v4i*)l, 0, 0);
#else
  *(v8h*)l = *(const v8h*)g;
#endif
}

__device__ __forceinline__ void wait_async_le8() {
#if USE_ASYNC_LDS
  __builtin_amdgcn_s_wait_asynccnt(8);
#endif
}
__device__ __forceinline__ void wait_async_le0() {
#if USE_ASYNC_LDS
  __builtin_amdgcn_s_wait_asynccnt(0);
#endif
}

// C(M,N) = A(M,K)*B(N,K)^T, all-f16 operands (K-contiguous), f32 accumulate.
// 128 threads = 4 wave32 waves in a 2x2 grid; each wave owns a 64x64 tile
// (4x4 WMMA frags). Double-buffered LDS, async global->LDS prefetch.
// Grid: (N/BN, M/BM, batches).
template <typename OutT, bool ACC>
__global__ __launch_bounds__(128) void gemm_f16_nt_kernel(
    const _Float16* __restrict__ A, const _Float16* __restrict__ B,
    OutT* __restrict__ C,
    int K, int lda, int ldb, int ldc,
    long long sA, long long sB, long long sC)
{
  __shared__ __align__(16) _Float16 ldsA[2][BM * LDK];
  __shared__ __align__(16) _Float16 ldsB[2][BN * LDK];

  A += (long long)blockIdx.z * sA;
  B += (long long)blockIdx.z * sB;
  C += (long long)blockIdx.z * sC;

  const int m0   = blockIdx.y * BM;
  const int n0   = blockIdx.x * BN;
  const int tid  = threadIdx.x;
  const int lane = tid & 31;
  const int wid  = tid >> 5;
  const int wm   = wid >> 1;     // 0..1 -> 64 rows
  const int wn   = wid & 1;      // 0..1 -> 64 cols
  const int r    = lane & 15;
  const int hh   = lane >> 4;

  // stage one 128x32 tile of A and of B into buffer `buf` (512+512 16B chunks)
  auto stage = [&](int buf, int k0) {
    #pragma unroll
    for (int i = 0; i < 4; ++i) {
      int c = tid + i * 128;
      int row = c >> 2;            // BK/8 = 4 chunks per row
      int kc  = (c & 3) * 8;
      copy16_g2l(A + (long long)(m0 + row) * lda + (k0 + kc), &ldsA[buf][row * LDK + kc]);
    }
    #pragma unroll
    for (int i = 0; i < 4; ++i) {
      int c = tid + i * 128;
      int row = c >> 2;
      int kc  = (c & 3) * 8;
      copy16_g2l(B + (long long)(n0 + row) * ldb + (k0 + kc), &ldsB[buf][row * LDK + kc]);
    }
  };

  v8f acc[4][4];
  #pragma unroll
  for (int i = 0; i < 4; ++i)
    #pragma unroll
    for (int j = 0; j < 4; ++j)
      #pragma unroll
      for (int e = 0; e < 8; ++e) acc[i][j][e] = 0.0f;

  const int nk = K / BK;
  stage(0, 0);

  for (int ik = 0; ik < nk; ++ik) {
    const int buf = ik & 1;
    if (ik + 1 < nk) {            // prefetch next tile into the other buffer
      stage(buf ^ 1, (ik + 1) * BK);
      wait_async_le8();           // the 8 just-issued may fly; tile ik is done
    } else {
      wait_async_le0();
    }
    __syncthreads();

    // fragment loads (ISA 16-bit A / B VGPR layouts)
    v16h af[4], bf[4];
    #pragma unroll
    for (int i = 0; i < 4; ++i) {
      const _Float16* pa = &ldsA[buf][(wm * 64 + i * 16 + r) * LDK + hh * 8];
      v8h lo = *(const v8h*)pa;          // K = hh*8 .. +8
      v8h hi = *(const v8h*)(pa + 16);   // K = 16+hh*8 .. +8
      af[i] = __builtin_shufflevector(lo, hi, 0,1,2,3,4,5,6,7,8,9,10,11,12,13,14,15);
    }
    #pragma unroll
    for (int j = 0; j < 4; ++j) {
      const _Float16* pb = &ldsB[buf][(wn * 64 + j * 16 + r) * LDK + hh * 8];
      v8h lo = *(const v8h*)pb;          // K = hh*8 .. +8
      v8h hi = *(const v8h*)(pb + 16);   // K = 16+hh*8 .. +8
      bf[j] = __builtin_shufflevector(lo, hi, 0,1,2,3,4,5,6,7,8,9,10,11,12,13,14,15);
    }

    #pragma unroll
    for (int i = 0; i < 4; ++i)
      #pragma unroll
      for (int j = 0; j < 4; ++j)
        acc[i][j] = __builtin_amdgcn_wmma_f32_16x16x32_f16(
            false, af[i], false, bf[j], (short)0, acc[i][j], false, false);
    __syncthreads();                // everyone done reading buf -> reusable
  }

  // store C: vgpr e -> row hh*8+e, lane r -> col
  #pragma unroll
  for (int i = 0; i < 4; ++i)
    #pragma unroll
    for (int j = 0; j < 4; ++j)
      #pragma unroll
      for (int e = 0; e < 8; ++e) {
        long long row = m0 + wm * 64 + i * 16 + hh * 8 + e;
        long long col = n0 + wn * 64 + j * 16 + r;
        OutT* dst = C + row * ldc + col;
        float v = acc[i][j][e];
        if constexpr (ACC) v += (float)*dst;
        *dst = (OutT)v;
      }
}

// ---------------- helpers ----------------
__global__ __launch_bounds__(256) void cvt_kernel(const float* __restrict__ in,
                                                  _Float16* __restrict__ out,
                                                  long long n) {
  long long i = (long long)blockIdx.x * 256 + threadIdx.x;
  if (i < n) out[i] = (_Float16)in[i];
}

// W_out (D, 2D) -> Wc (D,D) f16 (k<D) and Wt (D,D) f16 (k>=D)
__global__ __launch_bounds__(256) void split_wout_kernel(const float* __restrict__ w,
                                                         _Float16* __restrict__ wc,
                                                         _Float16* __restrict__ wt) {
  int i = blockIdx.x * 256 + threadIdx.x;   // over 1024*2048
  int d = i >> 11;
  int k = i & 2047;
  _Float16 v = (_Float16)w[i];
  if (k < 1024) wc[d * 1024 + k] = v;
  else          wt[d * 1024 + (k - 1024)] = v;
}

// src (b,S,D) f32 -> srcH (b,S,D) f16 AND srcT (b,D,S) f16 in one pass.
__global__ void transpose_cvt_kernel(const float* __restrict__ src,
                                     _Float16* __restrict__ dstH,
                                     _Float16* __restrict__ dstT,
                                     int S, int D) {
  __shared__ float tile[32][33];
  int b  = blockIdx.z;
  int s0 = blockIdx.x * 32;
  int d0 = blockIdx.y * 32;
  const float* sp = src  + (long long)b * S * D;
  _Float16*    hp = dstH + (long long)b * S * D;
  _Float16*    dp = dstT + (long long)b * D * S;
  int tx = threadIdx.x, ty = threadIdx.y;
  #pragma unroll
  for (int i = 0; i < 32; i += 8) {
    float v = sp[(long long)(s0 + ty + i) * D + d0 + tx];
    tile[ty + i][tx] = v;
    hp[(long long)(s0 + ty + i) * D + d0 + tx] = (_Float16)v;
  }
  __syncthreads();
  #pragma unroll
  for (int i = 0; i < 32; i += 8)
    dp[(long long)(d0 + ty + i) * S + s0 + tx] = (_Float16)tile[tx][ty + i];
}

// Row-wise masked softmax over S=2048: f32 in place + f16 mirror.
__global__ __launch_bounds__(256) void softmax_mask_kernel(float* __restrict__ score,
                                                           _Float16* __restrict__ scoreH,
                                                           const int* __restrict__ lens) {
  constexpr int S = 2048, ITER = S / 256;
  int b = blockIdx.y;
  long long row = (long long)b * gridDim.x + blockIdx.x;
  float*    p = score  + row * S;
  _Float16* q = scoreH + row * S;
  int len = lens[b];
  int tid = threadIdx.x;
  __shared__ float red[256];

  float v[ITER];
  float mx = -3.402823466e38f;
  #pragma unroll
  for (int i = 0; i < ITER; ++i) {
    int s = tid + i * 256;
    float x = (s < len) ? p[s] : -3.402823466e38f;
    v[i] = x;
    mx = fmaxf(mx, x);
  }
  red[tid] = mx; __syncthreads();
  for (int off = 128; off; off >>= 1) {
    if (tid < off) red[tid] = fmaxf(red[tid], red[tid + off]);
    __syncthreads();
  }
  mx = red[0]; __syncthreads();

  float sum = 0.0f;
  #pragma unroll
  for (int i = 0; i < ITER; ++i) {
    int s = tid + i * 256;
    float e = (s < len) ? __expf(v[i] - mx) : 0.0f;
    v[i] = e;
    sum += e;
  }
  red[tid] = sum; __syncthreads();
  for (int off = 128; off; off >>= 1) {
    if (tid < off) red[tid] += red[tid + off];
    __syncthreads();
  }
  float inv = 1.0f / red[0];
  #pragma unroll
  for (int i = 0; i < ITER; ++i) {
    float o = v[i] * inv;
    int s = tid + i * 256;
    p[s] = o;
    q[s] = (_Float16)o;
  }
}

// ---------------- launch ----------------
extern "C" void kernel_launch(void* const* d_in, const int* in_sizes, int n_in,
                              void* d_out, int out_size, void* d_ws, size_t ws_size,
                              hipStream_t stream) {
  (void)in_sizes; (void)n_in; (void)out_size; (void)ws_size;
  constexpr int BZ = 8, T = 2048, S = 2048, D = 1024;

  const float* src  = (const float*)d_in[0];
  const float* tgt  = (const float*)d_in[1];
  const float* Wg   = (const float*)d_in[2];
  const float* Wout = (const float*)d_in[3];
  const int*   lens = (const int*)d_in[4];

  float* attn  = (float*)d_out;                 // (BZ,T,D)
  float* align = attn + (long long)BZ * T * D;  // (BZ,T,S)

  char* w = (char*)d_ws;
  auto take = [&](size_t bytes) { char* p = w; w += (bytes + 255) & ~(size_t)255; return p; };
  _Float16* srcH   = (_Float16*)take((size_t)BZ * S * D * 2);  // src f16
  _Float16* srcT   = (_Float16*)take((size_t)BZ * D * S * 2);  // src^T f16
  _Float16* tgtH   = (_Float16*)take((size_t)BZ * T * D * 2);  // tgt f16
  _Float16* WgH    = (_Float16*)take((size_t)D * D * 2);
  _Float16* WcH    = (_Float16*)take((size_t)D * D * 2);
  _Float16* WtH    = (_Float16*)take((size_t)D * D * 2);
  _Float16* tgtpH  = (_Float16*)take((size_t)BZ * T * D * 2);  // tgt @ Wg^T
  _Float16* cH     = (_Float16*)take((size_t)BZ * T * D * 2);  // context
  _Float16* alignH = (_Float16*)take((size_t)BZ * T * S * 2);  // softmax f16

  // conversions
  cvt_kernel<<<(D * D) / 256, 256, 0, stream>>>(Wg, WgH, (long long)D * D);
  cvt_kernel<<<(BZ * T * D) / 256, 256, 0, stream>>>(tgt, tgtH, (long long)BZ * T * D);
  split_wout_kernel<<<(D * 2 * D) / 256, 256, 0, stream>>>(Wout, WcH, WtH);
  transpose_cvt_kernel<<<dim3(S / 32, D / 32, BZ), dim3(32, 8), 0, stream>>>(src, srcH, srcT, S, D);

  // 1) tgt_p = tgt @ Wg^T            (M=BZ*T, N=D, K=D)
  gemm_f16_nt_kernel<_Float16, false>
      <<<dim3(D / BN, (BZ * T) / BM, 1), 128, 0, stream>>>(
          tgtH, WgH, tgtpH, D, D, D, D, 0, 0, 0);

  // 2) score = tgt_p @ src^T         (batched: M=T, N=S, K=D) -> align f32
  gemm_f16_nt_kernel<float, false>
      <<<dim3(S / BN, T / BM, BZ), 128, 0, stream>>>(
          tgtpH, srcH, align, D, D, D, S,
          (long long)T * D, (long long)S * D, (long long)T * S);

  // 3) masked softmax (f32 output in d_out + f16 mirror for next GEMM)
  softmax_mask_kernel<<<dim3(T, BZ), 256, 0, stream>>>(align, alignH, lens);

  // 4) c = align @ src               (batched: M=T, N=D, K=S; B = srcT)
  gemm_f16_nt_kernel<_Float16, false>
      <<<dim3(D / BN, T / BM, BZ), 128, 0, stream>>>(
          alignH, srcT, cH, S, S, S, D,
          (long long)T * S, (long long)D * S, (long long)T * D);

  // 5) attn = c @ Wc^T ; attn += tgt @ Wt^T    (M=BZ*T, N=D, K=D)
  gemm_f16_nt_kernel<float, false>
      <<<dim3(D / BN, (BZ * T) / BM, 1), 128, 0, stream>>>(
          cH, WcH, attn, D, D, D, D, 0, 0, 0);
  gemm_f16_nt_kernel<float, true>
      <<<dim3(D / BN, (BZ * T) / BM, 1), 128, 0, stream>>>(
          tgtH, WtH, attn, D, D, D, D, 0, 0, 0);
}
